// PMS_25898652795297
// MI455X (gfx1250) — matrix operand order repaired and checked by reading, not compiled, not run on hardware
//
#include <hip/hip_runtime.h>
#include <math.h>

// ---------------- CDNA5 WMMA types ----------------
typedef __attribute__((ext_vector_type(16))) __bf16 v16bf;
typedef __attribute__((ext_vector_type(8)))  __bf16 v8bf;
typedef __attribute__((ext_vector_type(8)))  float  v8f;

#define LSTR 72           // LDS row stride (elements): 144B = 9*16 -> 16B-aligned rows, staggered banks
#define NPIX 16384        // 128*128
#define NTOT 4194304      // 4*64*128*128

__device__ __forceinline__ unsigned short f2bf(float f) {
  unsigned u = __builtin_bit_cast(unsigned, f);
  unsigned r = u + 0x7FFFu + ((u >> 16) & 1u);   // round-to-nearest-even
  return (unsigned short)(r >> 16);
}

// -------------------------------------------------------------------------
// Fused implicit-GEMM conv: 64ch -> 64ch, KSZ in {1,3,5}, B=4, H=W=128.
// Workgroup = (64-pixel strip) x (64 out ch) for one batch image; 8 waves,
// each wave: one 16-out-ch M tile x two 16-pixel N tiles. Double-buffered
// LDS staging (weights tap slice [o][k] + transposed input tile [pix][k],
// bf16) overlapped with v_wmma_f32_16x16x32_bf16. All WMMA fragments are
// built from pairs of 16B LDS loads (ds_load_b128). Compile-time epilogue:
// BIASM (0 none, 1 per-o, 2 torch repeat_interleave), ACT (0/1 relu/2 lrelu),
// HASRES (+resid), ACCUM (out +=).
// -------------------------------------------------------------------------
template <int KSZ, int BIASM, int ACT, bool HASCOND, bool HASRES, bool ACCUM>
__global__ __launch_bounds__(256) void conv_wmma_kernel(
    const float* __restrict__ in, const float* __restrict__ wgt,
    const float* __restrict__ bias, const float* __restrict__ cond,
    const float* __restrict__ resid, float* __restrict__ out, float wscale)
{
  constexpr int PAD  = (KSZ - 1) / 2;
  constexpr int TAPS = KSZ * KSZ;

  __shared__ unsigned short Wl[2][64 * LSTR];  // [out_ch][in_ch(K)]  bf16
  __shared__ unsigned short Xl[2][64 * LSTR];  // [pixel ][in_ch(K)]  bf16 (transposed)
  __shared__ float condS[64];

  const int wt   = blockIdx.x;          // 0..1  (64-pixel strip)
  const int h    = blockIdx.y;          // 0..127
  const int b    = blockIdx.z;          // 0..3
  const int w0   = wt * 64;
  const int tid  = threadIdx.x;
  const int lane = tid & 31;
  const int wv   = tid >> 5;
  const int mbase  = (wv & 3) * 16;     // out-channel tile base
  const int np0    = (wv >> 2) * 32;    // pixel tile base (two 16-wide tiles)
  const int lrow   = lane & 15;
  const int khalfA = (lane >> 4) * 8;   // unused; A fragment also uses khalfB
  const int khalfB = (lane >> 4) * 16;  // B-matrix K offset per lane half

  if (HASCOND) {
    if (tid < 64) condS[tid] = cond[b * 64 + tid];
  }

  // staging for one kernel tap into buffer `buf`
  auto stage = [&](int buf, int tap) {
    const int ky = tap / KSZ, kx = tap % KSZ;
    const int hy = h + ky - PAD;
    #pragma unroll
    for (int rep = 0; rep < 16; ++rep) {
      int idx = rep * 256 + tid;
      int o = idx >> 6, i = idx & 63;
      float f = wgt[((o * 64 + i) * KSZ + ky) * KSZ + kx] * wscale;
      Wl[buf][o * LSTR + i] = f2bf(f);
    }
    #pragma unroll
    for (int rep = 0; rep < 16; ++rep) {
      int idx = rep * 256 + tid;
      int i = idx >> 6, p = idx & 63;      // lane<->p: coalesced global reads
      int wx = w0 + p + kx - PAD;
      float v = 0.f;
      if (hy >= 0 && hy < 128 && wx >= 0 && wx < 128)
        v = in[((b * 64 + i) * 128 + hy) * 128 + wx];
      if (HASCOND) v *= condS[i];
      Xl[buf][p * LSTR + i] = f2bf(v);     // transposed store: [pixel][K]
    }
    // keep the weight stream ahead of the pipeline (global_prefetch_b8)
    if (tap + 2 < TAPS) {
      int t2 = tap + 2;
      int ky2 = t2 / KSZ, kx2 = t2 % KSZ;
      __builtin_prefetch(&wgt[((tid * 16) * KSZ + ky2) * KSZ + kx2], 0, 1);
    }
  };

  auto frag16 = [&](const unsigned short* base) -> v16bf {
    v8bf lo = *(const v8bf*)base;
    v8bf hi = *(const v8bf*)(base + 8);
    return __builtin_shufflevector(lo, hi, 0, 1, 2, 3, 4, 5, 6, 7,
                                   8, 9, 10, 11, 12, 13, 14, 15);
  };

  v8f acc0 = {}; v8f acc1 = {};

  if (HASCOND) __syncthreads();   // condS visible before first stage
  stage(0, 0);

  for (int t = 0; t < TAPS; ++t) {
    __syncthreads();
    if (t + 1 < TAPS) stage((t + 1) & 1, t + 1);
    const unsigned short* Wc = Wl[t & 1];
    const unsigned short* Xc = Xl[t & 1];

    #pragma unroll
    for (int cb = 0; cb < 64; cb += 32) {
      // A 16x32 bf16: elem e <-> K = khalfB + e (same contiguous order as B)
      const unsigned short* wrow = &Wc[(mbase + lrow) * LSTR + cb + khalfB];
      v8bf alo = *(const v8bf*)wrow;
      v8bf ahi = *(const v8bf*)(wrow + 8);
      v16bf af = __builtin_shufflevector(alo, ahi, 0, 1, 2, 3, 4, 5, 6, 7,
                                         8, 9, 10, 11, 12, 13, 14, 15);
      // B 32x16 bf16: elem e <-> K = khalfB + e -> 16 contiguous in [pix][K]
      v16bf bf0 = frag16(&Xc[(np0 + lrow) * LSTR + cb + khalfB]);
      v16bf bf1 = frag16(&Xc[(np0 + 16 + lrow) * LSTR + cb + khalfB]);

      acc0 = __builtin_amdgcn_wmma_f32_16x16x32_bf16(
               false, af, false, bf0, (short)0, acc0, false, false);
      acc1 = __builtin_amdgcn_wmma_f32_16x16x32_bf16(
               false, af, false, bf1, (short)0, acc1, false, false);
    }
  }

  // epilogue: C/D layout -> VGPR r, lane l: M = r + 8*(l>=16), N = l%16
  #pragma unroll
  for (int r = 0; r < 8; ++r) {
    int o = mbase + r + ((lane >> 4) * 8);
    float bv = 0.f;
    if (BIASM == 1) bv = bias[o];
    else if (BIASM == 2) bv = bias[(b * 64 + o) >> 2];  // repeat_interleave(b, B)
    #pragma unroll
    for (int t = 0; t < 2; ++t) {
      int n = np0 + t * 16 + lrow;
      float v = (t == 0 ? acc0[r] : acc1[r]) + bv;
      if (ACT == 1) v = fmaxf(v, 0.f);
      else if (ACT == 2) v = (v > 0.f) ? v : 0.01f * v;
      int idx = ((b * 64 + o) * 128 + h) * 128 + (w0 + n);
      if (HASRES) v += resid[idx];
      if (ACCUM) out[idx] += v;
      else       out[idx] = v;
    }
  }
}

// ---------------- adaptive max+avg pool to 1x1 over HxW -------------------
__global__ __launch_bounds__(256) void pool_kernel(
    const float* __restrict__ in, float* __restrict__ mx, float* __restrict__ av)
{
  __shared__ float smax[256];
  __shared__ float ssum[256];
  const int bc = blockIdx.x;                 // 0..255 = b*64+c
  const float* p = in + (size_t)bc * NPIX;
  float m = -INFINITY, s = 0.f;
  for (int i = threadIdx.x; i < NPIX; i += 256) { float v = p[i]; m = fmaxf(m, v); s += v; }
  smax[threadIdx.x] = m; ssum[threadIdx.x] = s;
  __syncthreads();
  for (int st = 128; st > 0; st >>= 1) {
    if (threadIdx.x < st) {
      smax[threadIdx.x] = fmaxf(smax[threadIdx.x], smax[threadIdx.x + st]);
      ssum[threadIdx.x] += ssum[threadIdx.x + st];
    }
    __syncthreads();
  }
  if (threadIdx.x == 0) { mx[bc] = smax[0]; av[bc] = ssum[0] * (1.f / NPIX); }
}

// ---- SE + FC: cse[b,o] = fw1*(fw0*sigmoid(se(mx)+se(av)) + fb0) + fb1 ----
__global__ void se_kernel(const float* __restrict__ mx, const float* __restrict__ av,
                          const float* __restrict__ cw1, const float* __restrict__ cw2,
                          const float* __restrict__ fw,  const float* __restrict__ fb,
                          float* __restrict__ cse)
{
  __shared__ float hm[4][4], ha[4][4];
  const int t = threadIdx.x;
  if (t < 32) {
    int src = t >> 4, b = (t >> 2) & 3, j = t & 3;
    const float* v = src ? av : mx;
    float s = 0.f;
    for (int c = 0; c < 64; ++c) s += cw1[j * 64 + c] * v[b * 64 + c];
    s = fmaxf(s, 0.f);
    if (src == 0) hm[b][j] = s; else ha[b][j] = s;
  }
  __syncthreads();
  int b = t >> 6, o = t & 63;
  float s = 0.f;
  for (int j = 0; j < 4; ++j) s += cw2[o * 4 + j] * (hm[b][j] + ha[b][j]);
  float x = 1.f / (1.f + expf(-s));
  cse[t] = fw[1] * (fw[0] * x + fb[0]) + fb[1];
}

// out[i] = cse[b*64+c] * acc[i]
__global__ void chscale_kernel(const float* __restrict__ acc,
                               const float* __restrict__ cse, float* __restrict__ out)
{
  int i = blockIdx.x * 256 + threadIdx.x;
  out[i] = cse[i >> 14] * acc[i];
}

// out[i] = a[i] + b[i]
__global__ void add_kernel(const float* __restrict__ a, const float* __restrict__ b,
                           float* __restrict__ out)
{
  int i = blockIdx.x * 256 + threadIdx.x;
  out[i] = a[i] + b[i];
}

// out[i] = x[i] * (1 + map[b,0,h,w])
__global__ void mapfma_kernel(const float* __restrict__ x, const float* __restrict__ map,
                              float* __restrict__ out)
{
  int i = blockIdx.x * 256 + threadIdx.x;
  int b = i >> 20;            // 64*128*128 = 2^20 elems per image
  int pix = i & (NPIX - 1);
  out[i] = x[i] * (1.f + map[b * NPIX + pix]);
}

// -------------------------------------------------------------------------
extern "C" void kernel_launch(void* const* d_in, const int* in_sizes, int n_in,
                              void* d_out, int out_size, void* d_ws, size_t ws_size,
                              hipStream_t stream)
{
  (void)in_sizes; (void)n_in; (void)out_size; (void)ws_size;
  const float* x      = (const float*)d_in[0];
  const float* c1     = (const float*)d_in[1];   // [9,2,4,64]
  const float* c2     = (const float*)d_in[2];   // [1,2,4,64]
  const float* mapP   = (const float*)d_in[3];   // [4,1,128,128]
  const float* res_w  = (const float*)d_in[4];   // [9,2,64,64,3,3]
  const float* res_b  = (const float*)d_in[5];   // [9,2,64]
  const float* mw1    = (const float*)d_in[6];   // [5,2,64,64,1,1]
  const float* mb1    = (const float*)d_in[7];
  const float* mw3    = (const float*)d_in[8];   // [5,2,64,64,3,3]
  const float* mb3    = (const float*)d_in[9];
  const float* mw5    = (const float*)d_in[10];  // [5,2,64,64,5,5]
  const float* mb5    = (const float*)d_in[11];
  const float* ca_w1  = (const float*)d_in[12];  // [5,4,64]
  const float* ca_w2  = (const float*)d_in[13];  // [5,64,4]
  const float* fc_w   = (const float*)d_in[14];  // [5,2]
  const float* fc_b   = (const float*)d_in[15];  // [5,2]

  float* ws = (float*)d_ws;
  const size_t N = NTOT;
  float* A   = ws + 0 * N;   // e1
  float* Bb  = ws + 1 * N;   // e2
  float* Cc  = ws + 2 * N;   // e3
  float* Dd  = ws + 3 * N;   // e4
  float* Ee  = ws + 4 * N;   // running tensor
  float* T   = ws + 5 * N;   // conv temp
  float* ACC = ws + 6 * N;   // MFE branch accumulator
  float* MX  = ws + 7 * N;
  float* AV  = MX + 256;
  float* CSE = MX + 512;

  const dim3 cgrid(2, 128, 4);
  const float SCALEK = 1.0f / 24.0f;   // 1/sqrt(64*3*3)
  const int EW = NTOT / 256;

  auto resblock = [&](int i, const float* in, float* out, const float* condbase) {
    const float* w0 = res_w + (size_t)(i * 2 + 0) * 64 * 64 * 9;
    const float* w1 = res_w + (size_t)(i * 2 + 1) * 64 * 64 * 9;
    const float* b0 = res_b + (i * 2 + 0) * 64;
    const float* b1 = res_b + (i * 2 + 1) * 64;
    // conv1: cond-modulated, bias repeat, relu
    conv_wmma_kernel<3, 2, 1, true, false, false><<<cgrid, 256, 0, stream>>>(
        in, w0, b0, condbase, nullptr, T, SCALEK);
    // conv2: cond-modulated, bias repeat, + residual(in)
    conv_wmma_kernel<3, 2, 0, true, true, false><<<cgrid, 256, 0, stream>>>(
        T, w1, b1, condbase + 256, in, out, SCALEK);
  };

  auto mfe = [&](int i, float* xbuf) {
    const float* w1a = mw1 + (size_t)(i * 2 + 0) * 4096;
    const float* w1b = mw1 + (size_t)(i * 2 + 1) * 4096;
    const float* w3a = mw3 + (size_t)(i * 2 + 0) * 36864;
    const float* w3b = mw3 + (size_t)(i * 2 + 1) * 36864;
    const float* w5a = mw5 + (size_t)(i * 2 + 0) * 102400;
    const float* w5b = mw5 + (size_t)(i * 2 + 1) * 102400;
    const float* b1a = mb1 + (i * 2) * 64, *b1b = mb1 + (i * 2 + 1) * 64;
    const float* b3a = mb3 + (i * 2) * 64, *b3b = mb3 + (i * 2 + 1) * 64;
    const float* b5a = mb5 + (i * 2) * 64, *b5b = mb5 + (i * 2 + 1) * 64;
    // 1x1 branch: lrelu then store into ACC
    conv_wmma_kernel<1, 1, 2, false, false, false><<<cgrid, 256, 0, stream>>>(
        xbuf, w1a, b1a, nullptr, nullptr, T, 1.f);
    conv_wmma_kernel<1, 1, 0, false, false, false><<<cgrid, 256, 0, stream>>>(
        T, w1b, b1b, nullptr, nullptr, ACC, 1.f);
    // 3x3 branch: lrelu then accumulate
    conv_wmma_kernel<3, 1, 2, false, false, false><<<cgrid, 256, 0, stream>>>(
        xbuf, w3a, b3a, nullptr, nullptr, T, 1.f);
    conv_wmma_kernel<3, 1, 0, false, false, true><<<cgrid, 256, 0, stream>>>(
        T, w3b, b3b, nullptr, nullptr, ACC, 1.f);
    // 5x5 branch: lrelu then accumulate
    conv_wmma_kernel<5, 1, 2, false, false, false><<<cgrid, 256, 0, stream>>>(
        xbuf, w5a, b5a, nullptr, nullptr, T, 1.f);
    conv_wmma_kernel<5, 1, 0, false, false, true><<<cgrid, 256, 0, stream>>>(
        T, w5b, b5b, nullptr, nullptr, ACC, 1.f);
    // channel attention + FC scale
    pool_kernel<<<256, 256, 0, stream>>>(xbuf, MX, AV);
    se_kernel<<<1, 256, 0, stream>>>(MX, AV, ca_w1 + i * 256, ca_w2 + i * 256,
                                     fc_w + i * 2, fc_b + i * 2, CSE);
    chscale_kernel<<<EW, 256, 0, stream>>>(ACC, CSE, xbuf);
  };

  // ---------------- network flow ----------------
  // e1
  resblock(0, x, A, c1 + 0 * 512);
  mfe(0, A);
  resblock(0, A, A, c1 + 0 * 512);
  // e2
  resblock(1, A,  Bb, c1 + 1 * 512);
  resblock(1, Bb, Bb, c1 + 1 * 512);
  resblock(1, Bb, Bb, c1 + 1 * 512);
  // e3
  resblock(2, Bb, Cc, c1 + 2 * 512);
  mfe(1, Cc);
  resblock(2, Cc, Cc, c1 + 2 * 512);
  // e4
  resblock(3, Cc, Dd, c1 + 3 * 512);
  resblock(3, Dd, Dd, c1 + 3 * 512);
  resblock(3, Dd, Dd, c1 + 3 * 512);
  // e5
  resblock(4, Dd, Ee, c2);
  mfe(2, Ee);
  mapfma_kernel<<<EW, 256, 0, stream>>>(Ee, mapP, Ee);   // e5*map + e5
  resblock(4, Ee, Ee, c2);
  add_kernel<<<EW, 256, 0, stream>>>(Ee, Dd, Ee);        // e5 += e4
  // d6
  resblock(5, Ee, Ee, c1 + 5 * 512);
  resblock(5, Ee, Ee, c1 + 5 * 512);
  resblock(5, Ee, Ee, c1 + 5 * 512);
  // d7 = e3 + d6
  add_kernel<<<EW, 256, 0, stream>>>(Cc, Ee, Ee);
  resblock(6, Ee, Ee, c1 + 6 * 512);
  mfe(3, Ee);
  resblock(6, Ee, Ee, c1 + 6 * 512);
  // d8 = e2 + d7
  add_kernel<<<EW, 256, 0, stream>>>(Bb, Ee, Ee);
  resblock(7, Ee, Ee, c1 + 7 * 512);
  resblock(7, Ee, Ee, c1 + 7 * 512);
  resblock(7, Ee, Ee, c1 + 7 * 512);
  // d9 = e1 + d8
  add_kernel<<<EW, 256, 0, stream>>>(A, Ee, Ee);
  resblock(8, Ee, Ee, c1 + 8 * 512);
  mfe(4, Ee);
  resblock(8, Ee, (float*)d_out, c1 + 8 * 512);   // final conv writes d_out
}
